// VAE_25297357373561
// MI455X (gfx1250) — matrix operand order repaired and checked by reading, not compiled, not run on hardware
//
#include <hip/hip_runtime.h>
#include <hip/hip_bf16.h>

typedef unsigned int u32;
typedef __attribute__((ext_vector_type(4)))  u32    u32x4;
typedef __attribute__((ext_vector_type(16))) __bf16 bf16x16;
typedef __attribute__((ext_vector_type(8)))  float  f32x8;

using bfr = __hip_bfloat16;

#define LDA 40   // 32 + 8 pad (bf16 elems): 80B rows -> conflict-free frag reads
#define LDB 40

union Frag { u32x4 u[2]; bf16x16 v; };

// ---------------------------------------------------------------------------
// zero page used to redirect OOB async loads (keeps EXEC uniform, no LDS prezero)
// ---------------------------------------------------------------------------
__global__ void k_zero(float* __restrict__ p, int n)
{
    int i = blockIdx.x * 256 + threadIdx.x;
    if (i < n) p[i] = 0.f;
}

// ---------------------------------------------------------------------------
// conv0: 1->256 channels, k=4 s=2, input 124x124 padded(+2) to 128, out 64x64.
// Tiny K (16), not worth WMMA. Writes NHWC bf16 with ReLU.
// ---------------------------------------------------------------------------
__global__ __launch_bounds__(256) void k_conv0(
    const float* __restrict__ x, const float* __restrict__ w,
    const float* __restrict__ bias, bfr* __restrict__ out)
{
    int pix = blockIdx.x;            // b*4096 + oh*64 + ow
    int c   = threadIdx.x;           // 0..255
    int b = pix >> 12, s = pix & 4095, oh = s >> 6, ow = s & 63;
    __shared__ float xs[16];
    if (c < 16) {
        int kh = c >> 2, kw = c & 3;
        int ih = oh * 2 + kh - 3;    // pad(2) + conv pad(1)
        int iw = ow * 2 + kw - 3;
        float v = 0.f;
        if (ih >= 0 && ih < 124 && iw >= 0 && iw < 124)
            v = x[((size_t)b * 124 + ih) * 124 + iw];
        xs[c] = v;
    }
    __syncthreads();
    float sum = bias[c];
    #pragma unroll
    for (int k = 0; k < 16; ++k) sum += xs[k] * w[c * 16 + k];
    out[(size_t)pix * 256 + c] = __float2bfloat16(fmaxf(sum, 0.f));
}

// ---------------------------------------------------------------------------
// Weight repack: OIHW f32 -> [O][(kh*KW+kw)*I + i] bf16 (matches im2col order)
// ---------------------------------------------------------------------------
__global__ void k_wconv(const float* __restrict__ w, bfr* __restrict__ out,
                        int O, int I, int KH, int KW)
{
    int i = blockIdx.x * 256 + threadIdx.x;
    int K = I * KH * KW;
    if (i >= O * K) return;
    int o = i / K, k = i - o * K;
    int ci = k % I, rem = k / I;
    int kw = rem % KW, kh = rem / KW;
    out[i] = __float2bfloat16(w[(((size_t)o * I + ci) * KH + kh) * KW + kw]);
}

__global__ void k_relu(const bfr* __restrict__ in, bfr* __restrict__ out, int n)
{
    int i = blockIdx.x * 256 + threadIdx.x;
    if (i < n) out[i] = __float2bfloat16(fmaxf(__bfloat162float(in[i]), 0.f));
}

// ---------------------------------------------------------------------------
// Implicit-GEMM conv via v_wmma_f32_16x16x32_bf16.
// Block: 256 thr = 8 waves; tile 128(M) x 64(N), BK=32. Wave = 32x32 outputs.
// Double-buffered LDS, staged with global_load_async_to_lds_b128 (ASYNCcnt):
// issue next tile's 3 async ops, s_wait_asynccnt 3 (in-order completion) so
// the older group is resident, barrier, WMMA on current buffer.
// OOB (conv padding / N<64 weight rows) redirect to a zeroed page.
// ---------------------------------------------------------------------------
__global__ __launch_bounds__(256) void k_conv_wmma(
    const bfr* __restrict__ Ain, const bfr* __restrict__ W,
    const float* __restrict__ bias, const bfr* __restrict__ resid,
    const bfr* __restrict__ zpage, void* __restrict__ Outv,
    int Hin, int Win, int Cin, int Hout, int Wout, int Cout,
    int KH, int KW, int stride, int pad, int hwShift, int woShift,
    int relu, int outF32)
{
    __shared__ bfr As[2][128 * LDA];
    __shared__ bfr Bs[2][64 * LDB];

    const int tid  = threadIdx.x;
    const int lane = tid & 31;
    const int wave = tid >> 5;
    const int wm   = wave & 3;     // 4 waves along M (32 rows each)
    const int wn   = wave >> 2;    // 2 waves along N (32 cols each)
    const int r16  = lane & 15;
    const int hi   = lane >> 4;

    const int mTile = blockIdx.x * 128;
    const int nTile = blockIdx.y * 64;
    const int K = Cin * KH * KW;

    // A-load coords fixed across K loop (thread t -> row t/2, half t&1)
    const int arow  = tid >> 1;
    const int ahalf = tid & 1;
    int m  = mTile + arow;
    int b  = m >> hwShift;
    int s  = m & ((1 << hwShift) - 1);
    int oh = s >> woShift;
    int ow = s & ((1 << woShift) - 1);
    int ih0 = oh * stride - pad;
    int iw0 = ow * stride - pad;
    const bfr* abase = Ain + (size_t)b * Hin * Win * Cin;

    // B-load coords (thread t -> weight row t/4, 8-elem quarter t&3)
    const int brow = tid >> 2;
    const int bq   = tid & 3;
    const bool bvalid = (nTile + brow) < Cout;
    const bfr* bptr = W + (size_t)(nTile + brow) * K + bq * 8;

    const unsigned alds0 = (unsigned)(uintptr_t)&As[0][arow * LDA + ahalf * 16];
    const unsigned alds1 = (unsigned)(uintptr_t)&As[1][arow * LDA + ahalf * 16];
    const unsigned blds0 = (unsigned)(uintptr_t)&Bs[0][brow * LDB + bq * 8];
    const unsigned blds1 = (unsigned)(uintptr_t)&Bs[1][brow * LDB + bq * 8];

    int c0 = 0, kh = 0, kw = 0;
    auto issue = [&](int buf) {
        int ih = ih0 + kh, iw = iw0 + kw;
        const bfr* asrc = (ih >= 0 && ih < Hin && iw >= 0 && iw < Win)
            ? (abase + ((size_t)ih * Win + iw) * Cin + c0 + ahalf * 16)
            : zpage;
        const bfr* bsrc = bvalid ? bptr : zpage;
        unsigned alds = buf ? alds1 : alds0;
        unsigned blds = buf ? blds1 : blds0;
        // IOFFSET applies to both LDS and global sides (ISA 10.async):
        // two b128 ops move this thread's contiguous 32B A slice, one for B.
        asm volatile(
            "global_load_async_to_lds_b128 %0, %2, off\n\t"
            "global_load_async_to_lds_b128 %0, %2, off offset:16\n\t"
            "global_load_async_to_lds_b128 %1, %3, off"
            :: "v"(alds), "v"(blds), "v"(asrc), "v"(bsrc)
            : "memory");
        bptr += 32;
        c0 += 32;
        if (c0 >= Cin) { c0 = 0; ++kw; if (kw >= KW) { kw = 0; ++kh; } }
    };

    f32x8 zero = {0.f,0.f,0.f,0.f,0.f,0.f,0.f,0.f};
    f32x8 acc[2][2];
    acc[0][0] = zero; acc[0][1] = zero; acc[1][0] = zero; acc[1][1] = zero;

    const int nK = K >> 5;
    issue(0);
    for (int kb = 0; kb < nK; ++kb) {
        const int cur = kb & 1;
        if (kb + 1 < nK) {
            issue(cur ^ 1);
            asm volatile("s_wait_asynccnt 0x3" ::: "memory");
        } else {
            asm volatile("s_wait_asynccnt 0x0" ::: "memory");
        }
        __syncthreads();   // all waves' tile `cur` resident

        // ---- fragments per ISA 7.12.2 lane layouts ----
        Frag fa[2], fb[2];
        #pragma unroll
        for (int mi = 0; mi < 2; ++mi) {
            const bfr* ap = &As[cur][(wm * 32 + mi * 16 + r16) * LDA + hi * 8];
            fa[mi].u[0] = *(const u32x4*)ap;          // K 0-7 / 8-15
            fa[mi].u[1] = *(const u32x4*)(ap + 16);   // K 16-23 / 24-31
        }
        #pragma unroll
        for (int ni = 0; ni < 2; ++ni) {
            const bfr* bp = &Bs[cur][(wn * 32 + ni * 16 + r16) * LDB + hi * 8];
            fb[ni].u[0] = *(const u32x4*)bp;          // K 0-7 / 8-15
            fb[ni].u[1] = *(const u32x4*)(bp + 16);   // K 16-23 / 24-31
        }
        #pragma unroll
        for (int mi = 0; mi < 2; ++mi)
            #pragma unroll
            for (int ni = 0; ni < 2; ++ni)
                acc[mi][ni] = __builtin_amdgcn_wmma_f32_16x16x32_bf16(
                    false, fa[mi].v, false, fb[ni].v,
                    (short)0, acc[mi][ni], false, false);

        __syncthreads();   // done reading `cur` before it is refilled next iter
    }

    // ---- epilogue: bias (+resid) (+relu), bf16 or f32 out ----
    #pragma unroll
    for (int mi = 0; mi < 2; ++mi) {
        #pragma unroll
        for (int ni = 0; ni < 2; ++ni) {
            int gcol = nTile + wn * 32 + ni * 16 + r16;
            if (gcol >= Cout) continue;
            float bv = bias[gcol];
            #pragma unroll
            for (int rr = 0; rr < 8; ++rr) {
                int grow = mTile + wm * 32 + mi * 16 + hi * 8 + rr;
                size_t off = (size_t)grow * Cout + gcol;
                float v = acc[mi][ni][rr] + bv;
                if (resid) v += __bfloat162float(resid[off]);
                if (relu)  v = fmaxf(v, 0.f);
                if (outF32) ((float*)Outv)[off] = v;
                else        ((bfr*)Outv)[off] = __float2bfloat16(v);
            }
        }
    }
}

// ---------------------------------------------------------------------------
// VQ: codebook norms, then per-row distance + argmin + gather (f32)
// ---------------------------------------------------------------------------
__global__ void k_codenorm(const float* __restrict__ cbk, float* __restrict__ cn)
{
    int j = blockIdx.x * 256 + threadIdx.x;
    if (j >= 512) return;
    float s = 0.f;
    for (int d = 0; d < 256; ++d) { float e = cbk[d * 512 + j]; s += e * e; }
    cn[j] = s;
}

__global__ __launch_bounds__(256) void k_vq(
    const float* __restrict__ z, const float* __restrict__ cbk,
    const float* __restrict__ cn, float* __restrict__ out)
{
    int m = blockIdx.x;              // b*16 + spatial
    int t = threadIdx.x;             // 0..255
    __shared__ float zs[256];
    __shared__ float dist[256];
    __shared__ int   didx[256];
    zs[t] = z[(size_t)m * 256 + t];
    __syncthreads();
    float best = 3.4e38f; int bi = 0;
    for (int j = t; j < 512; j += 256) {
        float dot = 0.f;
        for (int d = 0; d < 256; ++d) dot += zs[d] * cbk[d * 512 + j];
        float dd = cn[j] - 2.f * dot;     // |z|^2 constant per row
        if (dd < best) { best = dd; bi = j; }
    }
    dist[t] = best; didx[t] = bi;
    __syncthreads();
    for (int o = 128; o > 0; o >>= 1) {
        if (t < o && dist[t + o] < dist[t]) { dist[t] = dist[t + o]; didx[t] = didx[t + o]; }
        __syncthreads();
    }
    int idx = didx[0];
    int b = m >> 4, sp = m & 15;
    // out (B, 256*16): channel-major NCHW flatten
    out[(size_t)b * 4096 + t * 16 + sp] = cbk[t * 512 + idx];
}

// ---------------------------------------------------------------------------
extern "C" void kernel_launch(void* const* d_in, const int* in_sizes, int n_in,
                              void* d_out, int out_size, void* d_ws, size_t ws_size,
                              hipStream_t stream)
{
    (void)in_sizes; (void)n_in; (void)out_size; (void)ws_size;
    const float* x    = (const float*)d_in[0];
    const float* w0   = (const float*)d_in[1];  const float* b0   = (const float*)d_in[2];
    const float* w1   = (const float*)d_in[3];  const float* b1   = (const float*)d_in[4];
    const float* w2   = (const float*)d_in[5];  const float* b2   = (const float*)d_in[6];
    const float* w3   = (const float*)d_in[7];  const float* b3   = (const float*)d_in[8];
    const float* w4   = (const float*)d_in[9];  const float* b4   = (const float*)d_in[10];
    const float* wf   = (const float*)d_in[11]; const float* bfb  = (const float*)d_in[12];
    const float* rw0a = (const float*)d_in[13]; const float* rb0a = (const float*)d_in[14];
    const float* rw0b = (const float*)d_in[15]; const float* rb0b = (const float*)d_in[16];
    const float* rw1a = (const float*)d_in[17]; const float* rb1a = (const float*)d_in[18];
    const float* rw1b = (const float*)d_in[19]; const float* rb1b = (const float*)d_in[20];
    const float* wq   = (const float*)d_in[21]; const float* bq   = (const float*)d_in[22];
    const float* cbk  = (const float*)d_in[23];

    char* ws = (char*)d_ws; size_t off = 0;
    auto alloc = [&](size_t bytes) -> void* {
        void* p = ws + off; off = (off + bytes + 255) & ~(size_t)255; return p;
    };

    float* zpage = (float*)alloc(256);                       // zero page (64 f32)
    // repacked bf16 weights
    bfr* wc1  = (bfr*)alloc((size_t)512 * 4096 * 2);
    bfr* wc2  = (bfr*)alloc((size_t)512 * 8192 * 2);
    bfr* wc3  = (bfr*)alloc((size_t)512 * 8192 * 2);
    bfr* wc4  = (bfr*)alloc((size_t)512 * 8192 * 2);
    bfr* wcf  = (bfr*)alloc((size_t)512 * 4608 * 2);
    bfr* wr0a = (bfr*)alloc((size_t)32  * 4608 * 2);
    bfr* wr0b = (bfr*)alloc((size_t)512 * 32   * 2);
    bfr* wr1a = (bfr*)alloc((size_t)32  * 4608 * 2);
    bfr* wr1b = (bfr*)alloc((size_t)512 * 32   * 2);
    bfr* wqc  = (bfr*)alloc((size_t)256 * 512  * 2);
    // activations, NHWC bf16
    bfr* h0 = (bfr*)alloc((size_t)64 * 64 * 64 * 256 * 2);   // 134 MB (L2-resident)
    bfr* h1 = (bfr*)alloc((size_t)64 * 32 * 32 * 512 * 2);
    bfr* h2 = (bfr*)alloc((size_t)64 * 16 * 16 * 512 * 2);
    bfr* h3 = (bfr*)alloc((size_t)64 * 8  * 8  * 512 * 2);
    bfr* h4 = (bfr*)alloc((size_t)64 * 4  * 4  * 512 * 2);
    bfr* hA = (bfr*)alloc((size_t)1024 * 512 * 2);
    bfr* hB = (bfr*)alloc((size_t)1024 * 512 * 2);
    bfr* hC = (bfr*)alloc((size_t)1024 * 512 * 2);
    bfr* rb = (bfr*)alloc((size_t)1024 * 512 * 2);
    bfr* tb = (bfr*)alloc((size_t)1024 * 32  * 2);
    float* z  = (float*)alloc((size_t)1024 * 256 * 4);
    float* cn = (float*)alloc((size_t)512 * 4);

    k_zero<<<dim3(1), dim3(256), 0, stream>>>(zpage, 64);

    auto wrepack = [&](const float* src, bfr* dst, int O, int I, int KH, int KW) {
        int n = O * I * KH * KW;
        k_wconv<<<dim3((n + 255) / 256), dim3(256), 0, stream>>>(src, dst, O, I, KH, KW);
    };
    wrepack(w1, wc1, 512, 256, 4, 4);
    wrepack(w2, wc2, 512, 512, 4, 4);
    wrepack(w3, wc3, 512, 512, 4, 4);
    wrepack(w4, wc4, 512, 512, 4, 4);
    wrepack(wf, wcf, 512, 512, 3, 3);
    wrepack(rw0a, wr0a, 32, 512, 3, 3);
    wrepack(rw0b, wr0b, 512, 32, 1, 1);
    wrepack(rw1a, wr1a, 32, 512, 3, 3);
    wrepack(rw1b, wr1b, 512, 32, 1, 1);
    wrepack(wq, wqc, 256, 512, 1, 1);

    // conv0 (pad+4x4 s2 + relu) -> h0 NHWC bf16
    k_conv0<<<dim3(64 * 64 * 64), dim3(256), 0, stream>>>(x, w0, b0, h0);

    auto conv = [&](const bfr* A, const bfr* Wm, const float* bias, const bfr* resid,
                    void* out, int Hin, int Win, int Cin, int Hout, int Wout, int Cout,
                    int KH, int KW, int st, int pd, int relu, int of32) {
        int M = 64 * Hout * Wout;
        dim3 g(M / 128, (Cout + 63) / 64);
        k_conv_wmma<<<g, dim3(256), 0, stream>>>(A, Wm, bias, resid, (const bfr*)zpage,
            out, Hin, Win, Cin, Hout, Wout, Cout, KH, KW, st, pd,
            __builtin_ctz(Hout * Wout), __builtin_ctz(Wout), relu, of32);
    };

    // downsampling stack
    conv(h0, wc1, b1, nullptr, h1, 64, 64, 256, 32, 32, 512, 4, 4, 2, 1, 1, 0);
    conv(h1, wc2, b2, nullptr, h2, 32, 32, 512, 16, 16, 512, 4, 4, 2, 1, 1, 0);
    conv(h2, wc3, b3, nullptr, h3, 16, 16, 512,  8,  8, 512, 4, 4, 2, 1, 1, 0);
    conv(h3, wc4, b4, nullptr, h4,  8,  8, 512,  4,  4, 512, 4, 4, 2, 1, 1, 0);
    // final 3x3 (no relu)
    conv(h4, wcf, bfb, nullptr, hA, 4, 4, 512, 4, 4, 512, 3, 3, 1, 1, 0, 0);

    int nh = 1024 * 512;
    // residual block 0
    k_relu<<<dim3(nh / 256), dim3(256), 0, stream>>>(hA, rb, nh);
    conv(rb, wr0a, rb0a, nullptr, tb, 4, 4, 512, 4, 4, 32, 3, 3, 1, 1, 1, 0);
    conv(tb, wr0b, rb0b, hA,      hB, 4, 4, 32,  4, 4, 512, 1, 1, 1, 0, 0, 0);
    // residual block 1
    k_relu<<<dim3(nh / 256), dim3(256), 0, stream>>>(hB, rb, nh);
    conv(rb, wr1a, rb1a, nullptr, tb, 4, 4, 512, 4, 4, 32, 3, 3, 1, 1, 1, 0);
    conv(tb, wr1b, rb1b, hB,      hC, 4, 4, 32,  4, 4, 512, 1, 1, 1, 0, 0, 0);
    // final relu + pre-VQ 1x1 -> z (f32)
    k_relu<<<dim3(nh / 256), dim3(256), 0, stream>>>(hC, rb, nh);
    conv(rb, wqc, bq, nullptr, z, 4, 4, 512, 4, 4, 256, 1, 1, 1, 0, 0, 1);

    // VQ lookup + gather -> d_out (64, 4096) f32
    k_codenorm<<<dim3(2), dim3(256), 0, stream>>>(cbk, cn);
    k_vq<<<dim3(1024), dim3(256), 0, stream>>>(z, cbk, cn, (float*)d_out);
}
